// ScHetGNN_47777216201315
// MI455X (gfx1250) — matrix operand order repaired and checked by reading, not compiled, not run on hardware
//
#include <hip/hip_runtime.h>
#include <hip/hip_bf16.h>
#include <math.h>

#define NN 1024
#define SS 32
#define DD 128
#define HH 8

typedef __attribute__((ext_vector_type(8)))  __bf16 v8bf;
typedef __attribute__((ext_vector_type(16))) __bf16 v16bf;
typedef __attribute__((ext_vector_type(8)))  float  v8f;

static __device__ __forceinline__ v16bf mk16(const __bf16* lo, const __bf16* hi) {
  v8bf a = *(const v8bf*)lo;
  v8bf b = *(const v8bf*)hi;
  v16bf r;
#pragma unroll
  for (int i = 0; i < 8; ++i) { r[i] = a[i]; r[i + 8] = b[i]; }
  return r;
}

// Async DMA: global -> LDS, 16 bytes per lane, GV mode (SADDR=off). ASYNCcnt.
static __device__ __forceinline__ void async_b128(unsigned int lds_off,
                                                  const void* gaddr) {
  asm volatile("global_load_async_to_lds_b128 %0, %1, off"
               :: "v"(lds_off), "v"(gaddr)
               : "memory");
}
static __device__ __forceinline__ void wait_async0() {
  asm volatile("s_wait_asynccnt 0x0" ::: "memory");
}

#define APAD 136  // bf16 row stride: 272B -> 4-bank advance per row, conflict-free frag reads

// ---------------------------------------------------------------------------
// C[M,Nout] = act(A[M,128] @ W[Nout,128]^T + bias), bf16 WMMA, K fixed = 128.
// Block: 256 thr (8 waves), tile 128(M) x 64(N); wave tile 32x32 (2x2 WMMA).
// ACT: 0 none, 1 leaky(0.01), 2 relu
// ---------------------------------------------------------------------------
template <int ACT>
__global__ __launch_bounds__(256) void gemm_bias_k128(
    const float* __restrict__ A, const float* __restrict__ W,
    const float* __restrict__ bias, float* __restrict__ C, int M, int Nout) {
  __shared__ __attribute__((aligned(16))) __bf16 As[128 * APAD];
  __shared__ __attribute__((aligned(16))) __bf16 Ws[64 * APAD];
  const int tid = threadIdx.x, lane = tid & 31, w = tid >> 5;
  const int half = lane >> 4, l15 = lane & 15;
  const long row0 = (long)blockIdx.y * 128;
  const int col0 = blockIdx.x * 64;

  // stage A: 128x128 f32 -> bf16 (16 float4 per thread)
#pragma unroll
  for (int i = 0; i < 16; ++i) {
    int lin = tid + i * 256;            // float4 index in [0,4096)
    int r = lin >> 5, c4 = lin & 31;
    float4 v = *(const float4*)(A + (row0 + r) * DD + c4 * 4);
    __bf16* dst = &As[r * APAD + c4 * 4];
    dst[0] = (__bf16)v.x; dst[1] = (__bf16)v.y; dst[2] = (__bf16)v.z; dst[3] = (__bf16)v.w;
  }
  // stage W tile: 64x128 (W rows are exactly B columns since B = W^T)
#pragma unroll
  for (int i = 0; i < 8; ++i) {
    int lin = tid + i * 256;            // [0,2048)
    int r = lin >> 5, c4 = lin & 31;
    float4 v = *(const float4*)(W + (long)(col0 + r) * DD + c4 * 4);
    __bf16* dst = &Ws[r * APAD + c4 * 4];
    dst[0] = (__bf16)v.x; dst[1] = (__bf16)v.y; dst[2] = (__bf16)v.z; dst[3] = (__bf16)v.w;
  }
  __syncthreads();

  const int wm = (w >> 1) * 32, wn = (w & 1) * 32;
  v8f acc[2][2] = {};
#pragma unroll
  for (int kc = 0; kc < 4; ++kc) {
    const int kb = kc * 32;
    v16bf a[2], b[2];
#pragma unroll
    for (int mi = 0; mi < 2; ++mi) {    // A frag: m=l15, k = base8+{0..7,16..23}
      const __bf16* p = &As[(wm + mi * 16 + l15) * APAD + kb + half * 8];
      a[mi] = mk16(p, p + 16);
    }
#pragma unroll
    for (int ni = 0; ni < 2; ++ni) {    // B frag: n=l15, k = base8+{0..7,16..23}
      const __bf16* p = &Ws[(wn + ni * 16 + l15) * APAD + kb + half * 8];
      b[ni] = mk16(p, p + 16);
    }
#pragma unroll
    for (int mi = 0; mi < 2; ++mi)
#pragma unroll
      for (int ni = 0; ni < 2; ++ni)
        acc[mi][ni] = __builtin_amdgcn_wmma_f32_16x16x32_bf16(
            false, a[mi], false, b[ni], (short)0, acc[mi][ni], false, false);
  }

#pragma unroll
  for (int mi = 0; mi < 2; ++mi)
#pragma unroll
    for (int ni = 0; ni < 2; ++ni) {
      const int col = col0 + wn + ni * 16 + l15;
      const float bv = bias[col];
#pragma unroll
      for (int v = 0; v < 8; ++v) {
        long r = row0 + wm + mi * 16 + half * 8 + v;
        float x = acc[mi][ni][v] + bv;
        if (ACT == 1) x = (x >= 0.f) ? x : 0.01f * x;
        if (ACT == 2) x = (x > 0.f) ? x : 0.f;
        C[r * Nout + col] = x;
      }
    }
}

// ---------------------------------------------------------------------------
// Core MHA attention: Sq=1. One wave per (n,h), lane = s.
// Q: [N,128]; KV: [S*N,256] (K cols 0..127, V cols 128..255). out: [N,128]
// ---------------------------------------------------------------------------
__global__ __launch_bounds__(256) void attn_core_k(
    const float* __restrict__ Q, const float* __restrict__ KV,
    float* __restrict__ out) {
  const int tid = threadIdx.x, lane = tid & 31;
  const int p = blockIdx.x * 8 + (tid >> 5);
  const int n = p >> 3, h = p & 7;
  const float* kb = KV + ((long)lane * NN + n) * 256 + h * 16;
  const float* qb = Q + (long)n * DD + h * 16;
  float sc = 0.f;
#pragma unroll
  for (int d = 0; d < 16; ++d) sc += qb[d] * kb[d];
  sc *= 0.25f;
  float mx = sc;
#pragma unroll
  for (int m = 16; m >= 1; m >>= 1) mx = fmaxf(mx, __shfl_xor(mx, m, 32));
  float e = __expf(sc - mx), sum = e;
#pragma unroll
  for (int m = 16; m >= 1; m >>= 1) sum += __shfl_xor(sum, m, 32);
  const float pp = e / sum;
  float ov[16];
#pragma unroll
  for (int d = 0; d < 16; ++d) ov[d] = pp * kb[128 + d];
#pragma unroll
  for (int d = 0; d < 16; ++d)
#pragma unroll
    for (int m = 16; m >= 1; m >>= 1) ov[d] += __shfl_xor(ov[d], m, 32);
  if (lane == 0)
#pragma unroll
    for (int d = 0; d < 16; ++d) out[(long)n * DD + h * 16 + d] = ov[d];
}

// ---------------------------------------------------------------------------
// Self-MHA (S=32) attention + mean over queries. Wave per (n,h), lane = sq.
// QKV: [S*N,384]. out: [N,128]
// ---------------------------------------------------------------------------
__global__ __launch_bounds__(256) void attn_self32_mean(
    const float* __restrict__ QKV, float* __restrict__ out) {
  __shared__ float Kw[8][32][16];
  __shared__ float Vw[8][32][16];
  const int tid = threadIdx.x, lane = tid & 31, w = tid >> 5;
  const int p = blockIdx.x * 8 + w;
  const int n = p >> 3, h = p & 7;
  const float* base = QKV + ((long)lane * NN + n) * 384;
  float ql[16];
#pragma unroll
  for (int d = 0; d < 16; ++d) ql[d] = base[h * 16 + d];
#pragma unroll
  for (int d = 0; d < 16; ++d) Kw[w][lane][d] = base[128 + h * 16 + d];
#pragma unroll
  for (int d = 0; d < 16; ++d) Vw[w][lane][d] = base[256 + h * 16 + d];
  __syncthreads();
  float sc[32];
#pragma unroll
  for (int sk = 0; sk < 32; ++sk) {
    float d0 = 0.f;
#pragma unroll
    for (int d = 0; d < 16; ++d) d0 += ql[d] * Kw[w][sk][d];
    sc[sk] = d0 * 0.25f;
  }
  float mx = sc[0];
#pragma unroll
  for (int i = 1; i < 32; ++i) mx = fmaxf(mx, sc[i]);
  float sum = 0.f;
#pragma unroll
  for (int i = 0; i < 32; ++i) { sc[i] = __expf(sc[i] - mx); sum += sc[i]; }
  const float inv = 1.f / sum;
  float ov[16] = {};
#pragma unroll
  for (int sk = 0; sk < 32; ++sk) {
    const float pp = sc[sk] * inv;
#pragma unroll
    for (int d = 0; d < 16; ++d) ov[d] += pp * Vw[w][sk][d];
  }
#pragma unroll
  for (int d = 0; d < 16; ++d)
#pragma unroll
    for (int m = 16; m >= 1; m >>= 1) ov[d] += __shfl_xor(ov[d], m, 32);
  if (lane == 0)
#pragma unroll
    for (int d = 0; d < 16; ++d)
      out[(long)n * DD + h * 16 + d] = ov[d] * (1.0f / 32.0f);
}

// ---------------------------------------------------------------------------
// Comb MHA (seq=3) attention + mean over 3 queries. Thread per (n,h).
// ---------------------------------------------------------------------------
__global__ __launch_bounds__(256) void attn_comb3_mean(
    const float* __restrict__ QKV, float* __restrict__ out) {
  const int p = blockIdx.x * blockDim.x + threadIdx.x;
  const int n = p >> 3, h = p & 7;
  float q[3][16], k[3][16], vv[3][16];
#pragma unroll
  for (int s = 0; s < 3; ++s) {
    const float* base = QKV + ((long)s * NN + n) * 384;
#pragma unroll
    for (int d = 0; d < 16; ++d) {
      q[s][d] = base[h * 16 + d];
      k[s][d] = base[128 + h * 16 + d];
      vv[s][d] = base[256 + h * 16 + d];
    }
  }
  float ov[16] = {};
#pragma unroll
  for (int sq = 0; sq < 3; ++sq) {
    float sc[3];
#pragma unroll
    for (int sk = 0; sk < 3; ++sk) {
      float d0 = 0.f;
#pragma unroll
      for (int d = 0; d < 16; ++d) d0 += q[sq][d] * k[sk][d];
      sc[sk] = d0 * 0.25f;
    }
    float mx = fmaxf(sc[0], fmaxf(sc[1], sc[2]));
    float sum = 0.f;
#pragma unroll
    for (int sk = 0; sk < 3; ++sk) { sc[sk] = __expf(sc[sk] - mx); sum += sc[sk]; }
    const float inv = 1.f / sum;
#pragma unroll
    for (int sk = 0; sk < 3; ++sk)
#pragma unroll
      for (int d = 0; d < 16; ++d) ov[d] += sc[sk] * inv * vv[sk][d];
  }
#pragma unroll
  for (int d = 0; d < 16; ++d)
    out[(long)n * DD + h * 16 + d] = ov[d] * (1.0f / 3.0f);
}

// row-normalize [1024,128] -> bf16: wave per row
__global__ __launch_bounds__(256) void rownorm_bf16_k(const float* __restrict__ X,
                                                      __bf16* __restrict__ Y) {
  const int row = blockIdx.x * 8 + (threadIdx.x >> 5);
  const int lane = threadIdx.x & 31;
  float4 v = *(const float4*)(X + (long)row * DD + lane * 4);
  float ss = v.x * v.x + v.y * v.y + v.z * v.z + v.w * v.w;
#pragma unroll
  for (int m = 16; m >= 1; m >>= 1) ss += __shfl_xor(ss, m, 32);
  const float inv = rsqrtf(ss);
  __bf16* d = Y + (long)row * DD + lane * 4;
  d[0] = (__bf16)(v.x * inv); d[1] = (__bf16)(v.y * inv);
  d[2] = (__bf16)(v.z * inv); d[3] = (__bf16)(v.w * inv);
}

// bulk f32 -> bf16 (exact multiple of 4 elems per thread)
__global__ __launch_bounds__(256) void tobf16_k(const float* __restrict__ x,
                                                __bf16* __restrict__ y) {
  const size_t i = (size_t)blockIdx.x * blockDim.x + threadIdx.x;  // float4 idx
  float4 v = *(const float4*)(x + i * 4);
  __bf16* d = y + i * 4;
  d[0] = (__bf16)v.x; d[1] = (__bf16)v.y; d[2] = (__bf16)v.z; d[3] = (__bf16)v.w;
}

__global__ __launch_bounds__(256) void zero_k(float* __restrict__ p, int n) {
  int i = blockIdx.x * blockDim.x + threadIdx.x;
  if (i < n) p[i] = 0.f;
}

// ---------------------------------------------------------------------------
// Cosine-attention flash kernel, all-bf16 inner loop.
// Block = (n-tile of 128 rows, s). 8 waves, wave owns 16 query rows.
// Scores bounded by 1 -> softmax via exp(score-1), no online-max rescaling.
// Bn tile staged by async DMA (global_load_async_to_lds_b128, ASYNCcnt);
// V tile staged manually (needs transpose). Accumulate over s with atomics.
// ---------------------------------------------------------------------------
__global__ __launch_bounds__(256) void cos_flash(
    const __bf16* __restrict__ qnb,  // [N,128] normalized query, bf16
    const __bf16* __restrict__ kb,   // [S,N,128] keys, bf16 (normalized on load for score side)
    float* __restrict__ out_acc) {   // [N,128] zero-initialized accumulator
  __shared__ __attribute__((aligned(16))) __bf16 Qs[128 * APAD];  // reused as P
  __shared__ __attribute__((aligned(16))) __bf16 Bn[64 * APAD];
  __shared__ __attribute__((aligned(16))) __bf16 Vt[128 * 72];    // [d][m]
  const int tid = threadIdx.x, lane = tid & 31, w = tid >> 5;
  const int half = lane >> 4, l15 = lane & 15;
  const int s = blockIdx.y;
  const int n0 = blockIdx.x * 128;
  const __bf16* Kb = kb + (size_t)s * NN * DD;

  // stage normalized query-side rows Kn_s[n0..n0+127] (bf16 in, bf16 out)
  for (int r8 = 0; r8 < 16; ++r8) {
    const int row = w * 16 + r8;
    const __bf16* src = Kb + (size_t)(n0 + row) * DD + lane * 4;
    float x0 = (float)src[0], x1 = (float)src[1], x2 = (float)src[2], x3 = (float)src[3];
    float ss = x0 * x0 + x1 * x1 + x2 * x2 + x3 * x3;
#pragma unroll
    for (int m = 16; m >= 1; m >>= 1) ss += __shfl_xor(ss, m, 32);
    const float inv = rsqrtf(ss);
    __bf16* dst = &Qs[row * APAD + lane * 4];
    dst[0] = (__bf16)(x0 * inv); dst[1] = (__bf16)(x1 * inv);
    dst[2] = (__bf16)(x2 * inv); dst[3] = (__bf16)(x3 * inv);
  }
  __syncthreads();
  v16bf aQ[4];
#pragma unroll
  for (int kc = 0; kc < 4; ++kc) {
    const __bf16* p = &Qs[(w * 16 + l15) * APAD + kc * 32 + half * 8];
    aQ[kc] = mk16(p, p + 16);
  }
  __syncthreads();
  __bf16* Pw = &Qs[w * 16 * 72];  // wave-private [16][72] bf16

  v8f accO[8] = {};
  float lsum[8] = {};

  for (int mt = 0; mt < 16; ++mt) {
    const int m0 = mt * 64;
    if (mt < 15) __builtin_prefetch(Kb + (size_t)(m0 + 64) * DD, 0, 1);
    // async DMA: qn tile (64x128 bf16, row-major, padded) -> Bn
#pragma unroll
    for (int i = 0; i < 4; ++i) {
      int lin = tid + i * 256;          // 16B-chunk index in [0,1024)
      int r = lin >> 4, c = lin & 15;   // 16 chunks per 256B row
      unsigned int loff = (unsigned int)(size_t)&Bn[r * APAD + c * 8];
      async_b128(loff, qnb + (size_t)(m0 + r) * DD + c * 8);
    }
    // manual transposed staging of V tile (bf16 copy, no convert)
#pragma unroll
    for (int i = 0; i < 8; ++i) {
      int lin = tid + i * 256;  // [0,2048) groups of 4 elems
      int r = lin >> 5, c4 = lin & 31;
      const __bf16* u = Kb + (size_t)(m0 + r) * DD + c4 * 4;
      Vt[(c4 * 4 + 0) * 72 + r] = u[0];
      Vt[(c4 * 4 + 1) * 72 + r] = u[1];
      Vt[(c4 * 4 + 2) * 72 + r] = u[2];
      Vt[(c4 * 4 + 3) * 72 + r] = u[3];
    }
    wait_async0();
    __syncthreads();
    // scores: 16 rows x 64 cols per wave
    v8f sc[4] = {};
#pragma unroll
    for (int kc = 0; kc < 4; ++kc)
#pragma unroll
      for (int nt = 0; nt < 4; ++nt) {
        const __bf16* p = &Bn[(nt * 16 + l15) * APAD + kc * 32 + half * 8];
        v16bf b = mk16(p, p + 16);
        sc[nt] = __builtin_amdgcn_wmma_f32_16x16x32_bf16(
            false, aQ[kc], false, b, (short)0, sc[nt], false, false);
      }
    // exp(score-1), row-sum partials, stage P (C-layout -> LDS)
#pragma unroll
    for (int nt = 0; nt < 4; ++nt)
#pragma unroll
      for (int v = 0; v < 8; ++v) {
        float pv = __expf(sc[nt][v] - 1.0f);
        lsum[v] += pv;
        Pw[(half * 8 + v) * 72 + nt * 16 + l15] = (__bf16)pv;
      }
    // out += P @ V
#pragma unroll
    for (int kc = 0; kc < 2; ++kc) {
      const __bf16* pa = &Pw[l15 * 72 + kc * 32 + half * 8];
      v16bf aP = mk16(pa, pa + 16);
#pragma unroll
      for (int nt = 0; nt < 8; ++nt) {
        const __bf16* pb = &Vt[(nt * 16 + l15) * 72 + kc * 32 + half * 8];
        v16bf b = mk16(pb, pb + 16);
        accO[nt] = __builtin_amdgcn_wmma_f32_16x16x32_bf16(
            false, aP, false, b, (short)0, accO[nt], false, false);
      }
    }
    __syncthreads();
  }
  // finalize: reduce row sums over the 16-lane group, scale, atomic accumulate
#pragma unroll
  for (int v = 0; v < 8; ++v) {
    float l = lsum[v];
#pragma unroll
    for (int m = 8; m >= 1; m >>= 1) l += __shfl_xor(l, m, 32);
    const float rinv = 1.0f / l;
    const long rg = n0 + w * 16 + half * 8 + v;
#pragma unroll
    for (int nt = 0; nt < 8; ++nt)
      atomicAdd(&out_acc[rg * DD + nt * 16 + l15], accO[nt][v] * rinv);
  }
}

// ---------------------------------------------------------------------------
extern "C" void kernel_launch(void* const* d_in, const int* in_sizes, int n_in,
                              void* d_out, int out_size, void* d_ws, size_t ws_size,
                              hipStream_t stream) {
  const float* q_core     = (const float*)d_in[0];
  const float* neigh_core = (const float*)d_in[1];
  const float* neigh_key  = (const float*)d_in[2];
  const float* neigh_top  = (const float*)d_in[3];
  const float* core_in_w  = (const float*)d_in[4];
  const float* core_in_b  = (const float*)d_in[5];
  const float* core_out_w = (const float*)d_in[6];
  const float* core_out_b = (const float*)d_in[7];
  const float* key_in_w   = (const float*)d_in[8];
  const float* key_in_b   = (const float*)d_in[9];
  const float* key_out_w  = (const float*)d_in[10];
  const float* key_out_b  = (const float*)d_in[11];
  const float* top_in_w   = (const float*)d_in[12];
  const float* top_in_b   = (const float*)d_in[13];
  const float* top_out_w  = (const float*)d_in[14];
  const float* top_out_b  = (const float*)d_in[15];
  const float* comb_in_w  = (const float*)d_in[16];
  const float* comb_in_b  = (const float*)d_in[17];
  const float* comb_out_w = (const float*)d_in[18];
  const float* comb_out_b = (const float*)d_in[19];
  const float* pc_w = (const float*)d_in[20];
  const float* pc_b = (const float*)d_in[21];
  const float* pk_w = (const float*)d_in[22];
  const float* pk_b = (const float*)d_in[23];
  const float* pt_w = (const float*)d_in[24];
  const float* pt_b = (const float*)d_in[25];
  const float* fc_w = (const float*)d_in[26];
  const float* fc_b = (const float*)d_in[27];

  float* ws = (float*)d_ws;
  float* big     = ws;                                // [S*N, 384] = 48 MB
  float* t0      = big + (size_t)SS * NN * 384;       // [N,128]
  float* t1      = t0 + (size_t)NN * DD;
  float* t2      = t1 + (size_t)NN * DD;
  float* stacked = t2 + (size_t)NN * DD;              // [3N,128]
  float* qkvc    = stacked + (size_t)3 * NN * DD;     // [3N,384]
  float* tacc    = qkvc + (size_t)3 * NN * 384;       // [N,128]
  __bf16* kb16   = (__bf16*)(tacc + (size_t)NN * DD); // [S*N,128] bf16 = 8 MB
  __bf16* qnb    = kb16 + (size_t)SS * NN * DD;       // [N,128] bf16
  float* outp    = (float*)d_out;                     // [2,N,128]

  const dim3 blk(256);
  const dim3 gSmall(2, 8);     // Nout=128, M=1024

  // ---- core MHA (Sq=1) ----
  gemm_bias_k128<0><<<gSmall, blk, 0, stream>>>(q_core, core_in_w, core_in_b, t0, NN, 128);
  gemm_bias_k128<0><<<dim3(4, 256), blk, 0, stream>>>(neigh_core, core_in_w + 128 * 128,
                                                      core_in_b + 128, big, SS * NN, 256);
  attn_core_k<<<dim3(1024), blk, 0, stream>>>(t0, big, t1);
  gemm_bias_k128<0><<<gSmall, blk, 0, stream>>>(t1, core_out_w, core_out_b, t2, NN, 128);
  gemm_bias_k128<1><<<gSmall, blk, 0, stream>>>(t2, pc_w, pc_b, stacked, NN, 128);

  // ---- key MHA (self, S=32), mean commuted before out_proj ----
  gemm_bias_k128<0><<<dim3(6, 256), blk, 0, stream>>>(neigh_key, key_in_w, key_in_b, big, SS * NN, 384);
  attn_self32_mean<<<dim3(1024), blk, 0, stream>>>(big, t1);
  gemm_bias_k128<0><<<gSmall, blk, 0, stream>>>(t1, key_out_w, key_out_b, t2, NN, 128);
  gemm_bias_k128<1><<<gSmall, blk, 0, stream>>>(t2, pk_w, pk_b, stacked + NN * DD, NN, 128);

  // ---- top MHA ----
  gemm_bias_k128<0><<<dim3(6, 256), blk, 0, stream>>>(neigh_top, top_in_w, top_in_b, big, SS * NN, 384);
  attn_self32_mean<<<dim3(1024), blk, 0, stream>>>(big, t1);
  gemm_bias_k128<0><<<gSmall, blk, 0, stream>>>(t1, top_out_w, top_out_b, t2, NN, 128);
  gemm_bias_k128<1><<<gSmall, blk, 0, stream>>>(t2, pt_w, pt_b, stacked + 2 * NN * DD, NN, 128);

  // ---- comb MHA (seq=3), mean commuted ----
  gemm_bias_k128<0><<<dim3(6, 24), blk, 0, stream>>>(stacked, comb_in_w, comb_in_b, qkvc, 3 * NN, 384);
  attn_comb3_mean<<<dim3(32), blk, 0, stream>>>(qkvc, t1);
  gemm_bias_k128<0><<<gSmall, blk, 0, stream>>>(t1, comb_out_w, comb_out_b, outp, NN, 128);

  // ---- cosine attention (flash, bf16 inner loop) ----
  tobf16_k<<<dim3(4096), blk, 0, stream>>>(neigh_core, kb16);   // S*N*128/4 = 1M float4
  rownorm_bf16_k<<<dim3(128), blk, 0, stream>>>(q_core, qnb);
  zero_k<<<dim3(512), blk, 0, stream>>>(tacc, NN * DD);
  cos_flash<<<dim3(8, 32), blk, 0, stream>>>(qnb, kb16, tacc);
  gemm_bias_k128<2><<<gSmall, blk, 0, stream>>>(tacc, fc_w, fc_b, outp + NN * DD, NN, 128);
}
